// GNNBranch_89859305767799
// MI455X (gfx1250) — matrix operand order repaired and checked by reading, not compiled, not run on hardware
//
#include <hip/hip_runtime.h>
#include <math.h>

#define N_NODES 100000
#define N_EDGES 1600000
#define HID     128
#define NRBF    50
#define NLAYERS 3
#define NGRAPH  64
#define GAMMA   10.0f

typedef __attribute__((ext_vector_type(16))) _Float16 v16h;
typedef __attribute__((ext_vector_type(8)))  float    v8f;

__device__ __forceinline__ float silu_f(float x) { return x / (1.0f + __expf(-x)); }

// ---------------------------------------------------------------------------
// Pack f32 row-major weights [Krows x 128] into f16 WMMA B-fragments.
// Fragment memory layout: [kt][nt][lane][16 halves] so each lane loads its
// v16h with one contiguous 32-byte read.
// 16-bit B (32x16) VGPR layout: hi = lane>>4, elem e<8 -> K = 8*hi + e,
// elem e>=8 -> K = 16 + 8*hi + (e-8) ; N = nt*16 + (lane & 15).
// ---------------------------------------------------------------------------
__global__ void k_pack_bfrag(const float* __restrict__ W, _Float16* __restrict__ out,
                             int Krows, int KT) {
    int idx = blockIdx.x * 256 + threadIdx.x;
    int total = KT * 8 * 32 * 16;
    if (idx >= total) return;
    int e    = idx & 15;
    int lane = (idx >> 4) & 31;
    int nt   = (idx >> 9) & 7;
    int kt   = idx >> 12;
    int n = nt * 16 + (lane & 15);
    int k = kt * 32 + ((e < 8) ? (8 * (lane >> 4) + e) : (16 + 8 * (lane >> 4) + (e - 8)));
    float v = (k < Krows) ? W[k * HID + n] : 0.0f;
    out[idx] = (_Float16)v;
}

// h = silu(node_features @ emb_w + emb_b)
__global__ void k_node_embed(const float* __restrict__ nf, const float* __restrict__ w,
                             const float* __restrict__ b, float* __restrict__ h) {
    int t = blockIdx.x * blockDim.x + threadIdx.x;
    if (t >= N_NODES * HID) return;
    int n = t >> 7, c = t & 127;
    float acc = b[c];
    acc += nf[n * 3 + 0] * w[0 * HID + c];
    acc += nf[n * 3 + 1] * w[1 * HID + c];
    acc += nf[n * 3 + 2] * w[2 * HID + c];
    h[t] = silu_f(acc);
}

// ---------------------------------------------------------------------------
// Fused interaction layer: per 16-edge tile (one wave):
//   W = silu(rbf @ W1 + b1) @ W2 + b2   (WMMA f16 -> f32)
//   msg = h[src] * W ; atomicAdd(agg[dst], msg)
// 16-bit A (16x32) layout per lane: m = lane&15, hi = lane>>4,
//   elem e<8  -> k_local = 8*hi + e
//   elem e>=8 -> k_local = 16 + 8*hi + (e-8)
// ---------------------------------------------------------------------------
__global__ __launch_bounds__(256) void k_edge_interact(
    const float*    __restrict__ edge_dist,
    const int*      __restrict__ edge_index,   // [2,E]
    const float*    __restrict__ centers,      // [50]
    const _Float16* __restrict__ w1f,          // 2*8*32*16 (K padded 50->64)
    const float*    __restrict__ b1,           // [128]
    const _Float16* __restrict__ w2f,          // 4*8*32*16 (K=128)
    const float*    __restrict__ b2,           // [128]
    const float*    __restrict__ h,            // [N,128]
    float*          __restrict__ agg)          // [N,128]
{
    __shared__ _Float16 sA[8 * 16 * HID];      // per-wave 16x128 f16 staging (32 KB)
    int wave = threadIdx.x >> 5;               // wave32
    int lane = threadIdx.x & 31;
    int tile = blockIdx.x * 8 + wave;
    int ntiles = (N_EDGES + 15) >> 4;
    if (tile >= ntiles) return;                // wave-uniform; EXEC stays all-ones
    int ebase = tile * 16;
    int m  = lane & 15;
    int hi = lane >> 4;

    // ---- stage 1 A fragments: Gaussian RBF of this lane's edge row ----
    int eA = ebase + m;
    float d = (eA < N_EDGES) ? edge_dist[eA] : 0.0f;
    v16h a1[2];
#pragma unroll
    for (int kt = 0; kt < 2; ++kt) {
#pragma unroll
        for (int e = 0; e < 16; ++e) {
            int kk = (e < 8) ? (8 * hi + e) : (16 + 8 * hi + (e - 8));
            int k  = kt * 32 + kk;
            float v = 0.0f;
            if (k < NRBF) { float t = d - centers[k]; v = __expf(-GAMMA * t * t); }
            a1[kt][e] = (_Float16)v;
        }
    }

    // ---- stage 1 GEMM: acc[16x128] = rbf[16x64] @ W1[64x128] ----
    v8f acc[8] = {};
#pragma unroll
    for (int kt = 0; kt < 2; ++kt) {
#pragma unroll
        for (int nt = 0; nt < 8; ++nt) {
            v16h bf = *(const v16h*)(w1f + ((kt * 8 + nt) * 32 + lane) * 16);
            acc[nt] = __builtin_amdgcn_wmma_f32_16x16x32_f16(
                false, a1[kt], false, bf, (short)0, acc[nt], false, false);
        }
    }

    // ---- bias + silu -> LDS f16 row-major (same-wave region) ----
    // C/D layout: col n = nt*16 + (lane&15), row = r + 8*hi
    _Float16* my = sA + wave * 16 * HID;
#pragma unroll
    for (int nt = 0; nt < 8; ++nt) {
        int n = nt * 16 + m;
        float bb = b1[n];
#pragma unroll
        for (int r = 0; r < 8; ++r) {
            int row = r + 8 * hi;
            my[row * HID + n] = (_Float16)silu_f(acc[nt][r] + bb);
        }
    }
    asm volatile("s_wait_dscnt 0" ::: "memory");   // intra-wave LDS RAW fence

    // ---- stage 2 GEMM: acc2[16x128] = act[16x128] @ W2[128x128] ----
    v8f acc2[8] = {};
#pragma unroll
    for (int kt = 0; kt < 4; ++kt) {
        union { v16h h16; unsigned u32[8]; } a2;
#pragma unroll
        for (int p = 0; p < 8; ++p) {   // element pair (2p,2p+1) = 2 consecutive k
            int kk = (p < 4) ? (8 * hi + 2 * p) : (16 + 8 * hi + 2 * (p - 4));
            int k  = kt * 32 + kk;
            a2.u32[p] = *(const unsigned*)(my + m * HID + k);
        }
#pragma unroll
        for (int nt = 0; nt < 8; ++nt) {
            v16h bf = *(const v16h*)(w2f + ((kt * 8 + nt) * 32 + lane) * 16);
            acc2[nt] = __builtin_amdgcn_wmma_f32_16x16x32_f16(
                false, a2.h16, false, bf, (short)0, acc2[nt], false, false);
        }
    }

    // ---- epilogue: +b2, gather h[src], scatter-add agg[dst] (L2-resident) ----
    int srcs[8], dsts[8]; bool ok[8];
#pragma unroll
    for (int r = 0; r < 8; ++r) {
        int e = ebase + r + 8 * hi;
        ok[r]   = (e < N_EDGES);
        srcs[r] = ok[r] ? edge_index[e] : 0;
        dsts[r] = ok[r] ? edge_index[N_EDGES + e] : 0;
    }
#pragma unroll
    for (int nt = 0; nt < 8; ++nt) {
        int n = nt * 16 + m;
        float bb = b2[n];
#pragma unroll
        for (int r = 0; r < 8; ++r) {
            if (ok[r]) {
                float msg = (acc2[nt][r] + bb) * h[srcs[r] * HID + n];
                atomicAdd(agg + dsts[r] * HID + n, msg);
            }
        }
    }
}

// ---------------------------------------------------------------------------
// WMMA node update: h += silu(agg @ lin_w + lin_b)
// One wave per 16-node tile; A fragments built straight from global agg
// (consecutive-k f32 pairs -> packed f16), lin_w pre-packed B-fragments.
// Epilogue is a plain RMW (each element owned by one lane -> no atomics).
// ---------------------------------------------------------------------------
__global__ __launch_bounds__(256) void k_node_update_wmma(
    const float*    __restrict__ agg,
    const _Float16* __restrict__ lwf,   // 4*8*32*16
    const float*    __restrict__ lb,    // [128]
    float*          __restrict__ h)
{
    int wave = threadIdx.x >> 5;
    int lane = threadIdx.x & 31;
    int tile = blockIdx.x * 8 + wave;
    int ntiles = (N_NODES + 15) >> 4;          // 6250 (exact)
    if (tile >= ntiles) return;                // wave-uniform
    int nbase = tile * 16;
    int m  = lane & 15;
    int hi = lane >> 4;

    int arow_node = nbase + m;
    if (arow_node >= N_NODES) arow_node = N_NODES - 1;  // safe row (tail never hit: 100000 % 16 == 0)
    const float* arow = agg + (size_t)arow_node * HID;

    v8f acc[8] = {};
#pragma unroll
    for (int kt = 0; kt < 4; ++kt) {
        v16h a;
#pragma unroll
        for (int p = 0; p < 8; ++p) {          // element pair (2p,2p+1) = 2 consecutive k
            int kk = (p < 4) ? (8 * hi + 2 * p) : (16 + 8 * hi + 2 * (p - 4));
            int k  = kt * 32 + kk;
            float2 v2 = *(const float2*)(arow + k);
            a[2 * p]     = (_Float16)v2.x;
            a[2 * p + 1] = (_Float16)v2.y;
        }
#pragma unroll
        for (int nt = 0; nt < 8; ++nt) {
            v16h bf = *(const v16h*)(lwf + ((kt * 8 + nt) * 32 + lane) * 16);
            acc[nt] = __builtin_amdgcn_wmma_f32_16x16x32_f16(
                false, a, false, bf, (short)0, acc[nt], false, false);
        }
    }

#pragma unroll
    for (int nt = 0; nt < 8; ++nt) {
        int n = nt * 16 + m;
        float bb = lb[n];
#pragma unroll
        for (int r = 0; r < 8; ++r) {
            int node = nbase + r + 8 * hi;     // C/D row = r + 8*hi
            if (node < N_NODES) {
                size_t idx = (size_t)node * HID + n;
                h[idx] += silu_f(acc[nt][r] + bb);
            }
        }
    }
}

// gate = silu(h @ gw1 + gb1) @ gw2 + gb2   (one block of 64 threads per node)
__global__ __launch_bounds__(64) void k_gate(const float* __restrict__ h,
        const float* __restrict__ gw1, const float* __restrict__ gb1,
        const float* __restrict__ gw2, const float* __restrict__ gb2,
        float* __restrict__ gate) {
    __shared__ float red[64];
    int n = blockIdx.x, j = threadIdx.x;
    float acc = gb1[j];
    const float* hr = h + n * HID;
#pragma unroll 4
    for (int k = 0; k < HID; ++k) acc += hr[k] * gw1[k * 64 + j];
    red[j] = silu_f(acc) * gw2[j];
    __syncthreads();
    for (int s = 32; s > 0; s >>= 1) {
        if (j < s) red[j] += red[j + s];
        __syncthreads();
    }
    if (j == 0) gate[n] = red[0] + gb2[0];
}

// order-preserving uint encoding for float atomicMax
__device__ __forceinline__ unsigned enc_f(float f) {
    unsigned u = __float_as_uint(f);
    return (u & 0x80000000u) ? ~u : (u | 0x80000000u);
}
__device__ __forceinline__ float dec_f(unsigned u) {
    u = (u & 0x80000000u) ? (u & 0x7fffffffu) : ~u;
    return __uint_as_float(u);
}

__global__ void k_gate_max(const float* __restrict__ gate, const int* __restrict__ batch,
                           unsigned* __restrict__ gmax) {
    int n = blockIdx.x * blockDim.x + threadIdx.x;
    if (n >= N_NODES) return;
    atomicMax(gmax + batch[n], enc_f(gate[n]));
}

__global__ void k_gate_exp(const float* __restrict__ gate, const int* __restrict__ batch,
                           const unsigned* __restrict__ gmax,
                           float* __restrict__ ge, float* __restrict__ gsum) {
    int n = blockIdx.x * blockDim.x + threadIdx.x;
    if (n >= N_NODES) return;
    float v = __expf(gate[n] - dec_f(gmax[batch[n]]));
    ge[n] = v;
    atomicAdd(gsum + batch[n], v);
}

__global__ void k_pool(const float* __restrict__ h, const int* __restrict__ batch,
                       const float* __restrict__ ge, const float* __restrict__ gsum,
                       float* __restrict__ hg) {
    int t = blockIdx.x * blockDim.x + threadIdx.x;
    if (t >= N_NODES * HID) return;
    int n = t >> 7, c = t & 127;
    int b = batch[n];
    float alpha = ge[n] / (gsum[b] + 1e-8f);
    atomicAdd(hg + b * HID + c, alpha * h[t]);
}

// y = hg @ proj_w + proj_b ; LayerNorm ; exact GELU  (one block per graph)
__global__ __launch_bounds__(128) void k_final(const float* __restrict__ hg,
        const float* __restrict__ pw, const float* __restrict__ pb,
        const float* __restrict__ lg, const float* __restrict__ lbeta,
        float* __restrict__ out) {
    __shared__ float red[128];
    __shared__ float s_mu, s_rstd;
    int g = blockIdx.x, o = threadIdx.x;
    float acc = pb[o];
    const float* hr = hg + g * HID;
#pragma unroll 4
    for (int k = 0; k < HID; ++k) acc += hr[k] * pw[k * HID + o];
    red[o] = acc; __syncthreads();
    for (int s = 64; s > 0; s >>= 1) { if (o < s) red[o] += red[o + s]; __syncthreads(); }
    if (o == 0) s_mu = red[0] * (1.0f / 128.0f);
    __syncthreads();
    float dv = acc - s_mu;
    red[o] = dv * dv; __syncthreads();
    for (int s = 64; s > 0; s >>= 1) { if (o < s) red[o] += red[o + s]; __syncthreads(); }
    if (o == 0) s_rstd = rsqrtf(red[0] * (1.0f / 128.0f) + 1e-5f);
    __syncthreads();
    float y = dv * s_rstd * lg[o] + lbeta[o];
    out[g * HID + o] = 0.5f * y * (1.0f + erff(y * 0.70710678118654752f));
}

extern "C" void kernel_launch(void* const* d_in, const int* in_sizes, int n_in,
                              void* d_out, int out_size, void* d_ws, size_t ws_size,
                              hipStream_t stream) {
    const float* nf      = (const float*)d_in[0];
    const float* ed      = (const float*)d_in[1];
    const int*   ei      = (const int*)  d_in[2];
    const int*   batch   = (const int*)  d_in[3];
    const float* emb_w   = (const float*)d_in[4];
    const float* emb_b   = (const float*)d_in[5];
    const float* fw1     = (const float*)d_in[6];
    const float* fb1     = (const float*)d_in[7];
    const float* fw2     = (const float*)d_in[8];
    const float* fb2     = (const float*)d_in[9];
    const float* lw      = (const float*)d_in[10];
    const float* lb      = (const float*)d_in[11];
    const float* gw1     = (const float*)d_in[12];
    const float* gb1     = (const float*)d_in[13];
    const float* gw2     = (const float*)d_in[14];
    const float* gb2     = (const float*)d_in[15];
    const float* pw      = (const float*)d_in[16];
    const float* pb      = (const float*)d_in[17];
    const float* lng     = (const float*)d_in[18];
    const float* lnb     = (const float*)d_in[19];
    const float* centers = (const float*)d_in[20];

    char* ws = (char*)d_ws;
    size_t off = 0;
    auto alloc = [&](size_t bytes) -> char* {
        char* p = ws + off;
        off += (bytes + 255) & ~(size_t)255;
        return p;
    };
    float*    h    = (float*)alloc(sizeof(float) * N_NODES * HID);      // 51.2 MB
    float*    agg  = (float*)alloc(sizeof(float) * N_NODES * HID);      // 51.2 MB
    float*    gate = (float*)alloc(sizeof(float) * N_NODES);
    float*    ge   = (float*)alloc(sizeof(float) * N_NODES);
    unsigned* gmax = (unsigned*)alloc(sizeof(unsigned) * NGRAPH);
    float*    gsum = (float*)alloc(sizeof(float) * NGRAPH);
    float*    hg   = (float*)alloc(sizeof(float) * NGRAPH * HID);
    _Float16* w1f  = (_Float16*)alloc(sizeof(_Float16) * NLAYERS * 2 * 8 * 32 * 16);
    _Float16* w2f  = (_Float16*)alloc(sizeof(_Float16) * NLAYERS * 4 * 8 * 32 * 16);
    _Float16* lwf  = (_Float16*)alloc(sizeof(_Float16) * NLAYERS * 4 * 8 * 32 * 16);

    // pack filter + linear weights into WMMA B-fragment layout (f16)
    for (int l = 0; l < NLAYERS; ++l) {
        k_pack_bfrag<<<(2 * 8 * 32 * 16 + 255) / 256, 256, 0, stream>>>(
            fw1 + (size_t)l * NRBF * HID, w1f + (size_t)l * 2 * 8 * 32 * 16, NRBF, 2);
        k_pack_bfrag<<<(4 * 8 * 32 * 16 + 255) / 256, 256, 0, stream>>>(
            fw2 + (size_t)l * HID * HID, w2f + (size_t)l * 4 * 8 * 32 * 16, HID, 4);
        k_pack_bfrag<<<(4 * 8 * 32 * 16 + 255) / 256, 256, 0, stream>>>(
            lw + (size_t)l * HID * HID, lwf + (size_t)l * 4 * 8 * 32 * 16, HID, 4);
    }

    k_node_embed<<<(N_NODES * HID + 255) / 256, 256, 0, stream>>>(nf, emb_w, emb_b, h);

    int etiles = (N_EDGES + 15) / 16;           // 100000 -> 12500 full blocks
    int ntiles = (N_NODES + 15) / 16;           // 6250
    for (int l = 0; l < NLAYERS; ++l) {
        hipMemsetAsync(agg, 0, sizeof(float) * N_NODES * HID, stream);
        k_edge_interact<<<(etiles + 7) / 8, 256, 0, stream>>>(
            ed, ei, centers,
            w1f + (size_t)l * 2 * 8 * 32 * 16, fb1 + (size_t)l * HID,
            w2f + (size_t)l * 4 * 8 * 32 * 16, fb2 + (size_t)l * HID,
            h, agg);
        k_node_update_wmma<<<(ntiles + 7) / 8, 256, 0, stream>>>(
            agg, lwf + (size_t)l * 4 * 8 * 32 * 16, lb + (size_t)l * HID, h);
    }

    k_gate<<<N_NODES, 64, 0, stream>>>(h, gw1, gb1, gw2, gb2, gate);
    hipMemsetAsync(gmax, 0, sizeof(unsigned) * NGRAPH, stream);   // 0 == encoded minimum
    hipMemsetAsync(gsum, 0, sizeof(float) * NGRAPH, stream);
    hipMemsetAsync(hg, 0, sizeof(float) * NGRAPH * HID, stream);
    k_gate_max<<<(N_NODES + 255) / 256, 256, 0, stream>>>(gate, batch, gmax);
    k_gate_exp<<<(N_NODES + 255) / 256, 256, 0, stream>>>(gate, batch, gmax, ge, gsum);
    k_pool<<<(N_NODES * HID + 255) / 256, 256, 0, stream>>>(h, batch, ge, gsum, hg);
    k_final<<<NGRAPH, 128, 0, stream>>>(hg, pw, pb, lng, lnb, (float*)d_out);
}
